// CurricularFace_747324310282
// MI455X (gfx1250) — matrix operand order repaired and checked by reading, not compiled, not run on hardware
//
#include <hip/hip_runtime.h>
#include <hip/hip_bf16.h>

// Problem constants (fixed by the reference)
#define NROWS 512
#define KDIM  512
#define NCOLS 100000

#define COS_M_F    0.8775825618903728f   // cos(0.5)
#define SIN_M_F    0.4794255386042030f   // sin(0.5)
#define THRESH_F  (-0.8775825618903728f) // cos(pi - 0.5)
#define MM_F       0.2397127693021015f   // sin(pi - 0.5) * 0.5
#define SCALE_F    64.0f
#define EPS_F      1e-12f

typedef __attribute__((ext_vector_type(16))) __bf16       v16bf;
typedef __attribute__((ext_vector_type(8)))  float        v8f;
typedef __attribute__((ext_vector_type(4)))  unsigned int v4u;
typedef __attribute__((ext_vector_type(4)))  int          v4i;
typedef __attribute__((ext_vector_type(8)))  int          v8i;

union BFrag { v4u u[2]; v16bf v; };

#if defined(__HIP_DEVICE_COMPILE__) && __has_builtin(__builtin_amdgcn_tensor_load_to_lds)
#define HAVE_TDM 1
#else
#define HAVE_TDM 0
#endif

__device__ __forceinline__ unsigned short f32_to_bf16_rne(float f) {
    unsigned int u = __float_as_uint(f);
    unsigned int r = u + 0x7FFFu + ((u >> 16) & 1u);
    return (unsigned short)(r >> 16);
}

#if HAVE_TDM
// Issue a TDM 2D tile load: [rows x 64] fp32 tile at gptr (row stride NCOLS
// floats) -> LDS at lds_off. OOB columns/rows (beyond rem_cols/rem_rows) are
// zero-filled by the TDM, which replaces the column tail guard.
__device__ __forceinline__ void tdm_load_tile(const float* gptr, unsigned lds_off,
                                              unsigned rem_cols, unsigned rem_rows) {
    unsigned long long ga = (unsigned long long)(uintptr_t)gptr;
    v4u g0;
    g0[0] = 1u;                                        // count=1, user load desc
    g0[1] = lds_off;                                   // lds_addr (bytes)
    g0[2] = (unsigned)(ga & 0xFFFFFFFFull);            // global_addr[31:0]
    g0[3] = (unsigned)((ga >> 32) & 0x01FFFFFFull)     // global_addr[56:32]
          | (2u << 30);                                // type=2 ("image")
    v8i g1;
    g1[0] = (int)(2u << 16);                           // data_size=2 (4 bytes)
    g1[1] = (int)((rem_cols & 0xFFFFu) << 16);         // tensor_dim0[15:0]
    g1[2] = (int)((rem_cols >> 16) | ((rem_rows & 0xFFFFu) << 16)); // td0 hi | td1 lo
    g1[3] = (int)((rem_rows >> 16) | (64u << 16));     // td1 hi | tile_dim0=64
    g1[4] = (int)32u;                                  // tile_dim1=32, tile_dim2=0
    g1[5] = (int)(unsigned)NCOLS;                      // tensor_dim0_stride[31:0]
    g1[6] = 0;                                         // stride hi | dim1_stride lo
    g1[7] = 0;
    v4i z4 = {};
#if defined(__clang_major__) && (__clang_major__ >= 23)
    v8i z8 = {};
    __builtin_amdgcn_tensor_load_to_lds(g0, g1, z4, z4, z8, 0);
#else
    __builtin_amdgcn_tensor_load_to_lds(g0, g1, z4, z4, 0);
#endif
}
#endif

// ---------------- Kernel 1: embeddings fp32 -> bf16 (one-time, 1 MB read) ----
__global__ void cvt_emb_kernel(const float* __restrict__ emb,
                               unsigned short* __restrict__ abf) {
    int i = blockIdx.x * blockDim.x + threadIdx.x;   // 1024 * 256 = 262144
    abf[i] = f32_to_bf16_rne(emb[i]);
}

// ---------------- Kernel 2: per-row target logit (exact fp32) ---------------
__global__ void target_logit_kernel(const float* __restrict__ emb,
                                    const int*   __restrict__ label,
                                    const float* __restrict__ kern,
                                    float* __restrict__ ws_target) {
    __shared__ float sd[256];
    __shared__ float ss[256];
    const int n = blockIdx.x;          // row
    const int t = threadIdx.x;         // 256 threads
    const int lb = label[n];
    float d = 0.0f, s = 0.0f;
    for (int k = t; k < KDIM; k += 256) {
        float kv = kern[(size_t)k * NCOLS + lb];
        d += emb[n * KDIM + k] * kv;
        s += kv * kv;
    }
    sd[t] = d; ss[t] = s;
    __syncthreads();
    for (int o = 128; o > 0; o >>= 1) {
        if (t < o) { sd[t] += sd[t + o]; ss[t] += ss[t + o]; }
        __syncthreads();
    }
    if (t == 0) {
        float tl = sd[0] / fmaxf(sqrtf(ss[0]), EPS_F);
        ws_target[n] = fminf(fmaxf(tl, -1.0f), 1.0f);
    }
}

// ---------------- Kernel 3: t_new, cos_theta_m, final_target_logit ----------
__global__ void prep_kernel(const float* __restrict__ ws_target,
                            const float* __restrict__ t_in,
                            float* __restrict__ ws_ctm,
                            float* __restrict__ ws_ftl,
                            float* __restrict__ ws_tnew) {
    __shared__ float sm[NROWS];
    const int t = threadIdx.x;         // 512 threads, 1 block
    float tl = ws_target[t];
    sm[t] = tl;
    __syncthreads();
    for (int o = 256; o > 0; o >>= 1) {
        if (t < o) sm[t] += sm[t + o];
        __syncthreads();
    }
    if (t == 0) ws_tnew[0] = sm[0] * (1.0f / NROWS) * 0.01f + 0.99f * t_in[0];
    float st  = sqrtf(fmaxf(0.0f, 1.0f - tl * tl));
    float ctm = tl * COS_M_F - st * SIN_M_F;
    ws_ctm[t] = ctm;
    ws_ftl[t] = (tl > THRESH_F) ? ctm : (tl - MM_F);
}

// ---------------- Kernel 4: fused normalized GEMM + CurricularFace epilogue -
// Grid.x = column strips of 64; block = 512 threads = 16 wave32.
// Each workgroup covers all 512 output rows -> kernel matrix streamed from
// DRAM exactly once. B tile arrives via double-buffered TDM tensor_load_to_lds
// (fp32), is converted to bf16 into a transposed WMMA-layout LDS tile, and the
// per-column sum-of-squares for normalization is fused into the conversion.
#define BN   64
#define KB   32
#define LPIT 40   // LDS B pitch in bf16 elems: 80 B row, 16 B aligned, 20-bank stride

__global__ __launch_bounds__(512, 1)
void gemm_curricular_kernel(const float* __restrict__ kern,
                            const int*   __restrict__ label,
                            const unsigned short* __restrict__ abf,
                            const float* __restrict__ ws_ctm,
                            const float* __restrict__ ws_ftl,
                            const float* __restrict__ ws_tnew,
                            float* __restrict__ out) {
#if HAVE_TDM
    __shared__ __align__(16) float ldsRaw[2][KB * BN];       // 2 x 8 KB fp32 ping-pong
#endif
    __shared__ __align__(16) unsigned short ldsB[BN * LPIT]; // transposed bf16 B tile
    __shared__ float ssArr[512];
    __shared__ float colss[BN];
    __shared__ float ldsCtm[NROWS];
    __shared__ float ldsFtl[NROWS];
    __shared__ int   ldsLbl[NROWS];

    const int t    = threadIdx.x;
    const int w    = t >> 5;                 // wave id 0..15
    const int lane = t & 31;
    const int lo16 = lane & 15;
    const int hi   = (lane >> 4) & 1;        // 0: lanes 0-15, 1: lanes 16-31
    const int c0   = blockIdx.x * BN;

    // stage per-row epilogue params
    ldsCtm[t] = ws_ctm[t];
    ldsFtl[t] = ws_ftl[t];
    ldsLbl[t] = label[t];
    const float tnew = ws_tnew[0];

    const int n  = t & 63;                   // my staged column (fixed all K)
    const int kb = t >> 6;                   // my base K row within tile

    v8f acc[2][4];
    #pragma unroll
    for (int mi = 0; mi < 2; ++mi)
        #pragma unroll
        for (int ni = 0; ni < 4; ++ni) {
            v8f z = {};
            acc[mi][ni] = z;
        }
    float ssq = 0.0f;                        // partial sum of squares, column n

#if HAVE_TDM
    const unsigned rawOff0 = (unsigned)(uintptr_t)(const void*)&ldsRaw[0][0];
    const unsigned rawOff1 = (unsigned)(uintptr_t)(const void*)&ldsRaw[1][0];
    const unsigned remCols = (unsigned)(NCOLS - c0);
    if (w == 0)                               // prime the pipeline: tile 0
        tdm_load_tile(kern + c0, rawOff0, remCols, (unsigned)KDIM);
#endif

    for (int k0 = 0; k0 < KDIM; k0 += KB) {
#if HAVE_TDM
        const int cur = (k0 / KB) & 1;
        if (w == 0) {
            __builtin_amdgcn_s_wait_tensorcnt(0);   // tile k0 resident in LDS
            if (k0 + KB < KDIM)                     // launch tile k0+KB (async)
                tdm_load_tile(kern + (size_t)(k0 + KB) * NCOLS + c0,
                              cur ? rawOff0 : rawOff1,
                              remCols, (unsigned)(KDIM - (k0 + KB)));
        }
        __syncthreads();   // raw[cur] visible to all; prev ldsB readers done
        // convert fp32 tile -> bf16 transposed WMMA layout, fuse column sumsq
        #pragma unroll
        for (int i = 0; i < 4; ++i) {
            int kk  = kb + 8 * i;
            float v = ldsRaw[cur][kk * BN + n];     // OOB cols already zero
            ssq += v * v;
            ldsB[n * LPIT + kk] = f32_to_bf16_rne(v);
        }
        __syncthreads();
#else
        if (k0 + KB < KDIM)
            __builtin_prefetch(&kern[(size_t)(k0 + KB + kb) * NCOLS + c0 + n], 0, 1);
        __syncthreads();
        // branchless staging: clamped in-bounds load + select, no exec-mask dance
        #pragma unroll
        for (int i = 0; i < 4; ++i) {
            int kk   = kb + 8 * i;
            int col  = c0 + n;
            int colc = col < (NCOLS - 1) ? col : (NCOLS - 1);
            float v  = kern[(size_t)(k0 + kk) * NCOLS + colc];
            v = (col < NCOLS) ? v : 0.0f;
            ssq += v * v;
            ldsB[n * LPIT + kk] = f32_to_bf16_rne(v);
        }
        __syncthreads();
#endif

        // A fragments (bf16, 16x32): lanes 0-15 hold K 0..7 & 16..23 of row M=lane
        BFrag a[2], b[4];
        const int kA = k0 + (hi ? 8 : 0);
        #pragma unroll
        for (int mi = 0; mi < 2; ++mi) {
            int r = 32 * w + mi * 16 + lo16;
            const unsigned short* p = abf + r * KDIM + kA;
            a[mi].u[0] = *(const v4u*)p;
            a[mi].u[1] = *(const v4u*)(p + 16);
        }
        // B fragments (bf16, 32x16): lane = column, K 0..7 & 16..23 (lanes 16-31: +8)
        const int kB = hi ? 8 : 0;
        #pragma unroll
        for (int ni = 0; ni < 4; ++ni) {
            const unsigned short* p = &ldsB[(ni * 16 + lo16) * LPIT + kB];
            b[ni].u[0] = *(const v4u*)p;
            b[ni].u[1] = *(const v4u*)(p + 16);
        }
        #pragma unroll
        for (int mi = 0; mi < 2; ++mi)
            #pragma unroll
            for (int ni = 0; ni < 4; ++ni)
                acc[mi][ni] = __builtin_amdgcn_wmma_f32_16x16x32_bf16(
                    false, a[mi].v, false, b[ni].v,
                    (short)0, acc[mi][ni], false, false);
    }

    // deterministic column sum-of-squares reduction (8 partials per column)
    ssArr[t] = ssq;
    __syncthreads();
    if (t < BN) {
        float s = 0.0f;
        #pragma unroll
        for (int j = 0; j < 8; ++j) s += ssArr[t + 64 * j];
        colss[t] = s;
    }
    __syncthreads();

    float invn[4];
    #pragma unroll
    for (int ni = 0; ni < 4; ++ni)
        invn[ni] = 1.0f / fmaxf(sqrtf(colss[ni * 16 + lo16]), EPS_F);

    // epilogue: normalize, clip, hard-example reweight, label scatter, scale
    #pragma unroll
    for (int mi = 0; mi < 2; ++mi) {
        #pragma unroll
        for (int ni = 0; ni < 4; ++ni) {
            int c = c0 + ni * 16 + lo16;
            #pragma unroll
            for (int e = 0; e < 8; ++e) {
                int r = 32 * w + mi * 16 + e + (hi ? 8 : 0);
                float cosv = acc[mi][ni][e] * invn[ni];
                cosv = fminf(fmaxf(cosv, -1.0f), 1.0f);
                float v = (cosv > ldsCtm[r]) ? cosv * (tnew + cosv) : cosv;
                if (c == ldsLbl[r]) v = ldsFtl[r];
                if (c < NCOLS) out[(size_t)r * NCOLS + c] = SCALE_F * v;
            }
        }
    }
}

// ---------------- Launcher ---------------------------------------------------
extern "C" void kernel_launch(void* const* d_in, const int* in_sizes, int n_in,
                              void* d_out, int out_size, void* d_ws, size_t ws_size,
                              hipStream_t stream) {
    const float* emb   = (const float*)d_in[0];
    const int*   label = (const int*)  d_in[1];
    const float* kern  = (const float*)d_in[2];
    const float* t_in  = (const float*)d_in[3];
    float*       out   = (float*)d_out;

    char* ws = (char*)d_ws;
    unsigned short* abf = (unsigned short*)ws;                 // 512*512*2 = 524288 B
    float* ws_target = (float*)(ws + 524288);                  // 512 f
    float* ws_ctm    = ws_target + NROWS;                      // 512 f
    float* ws_ftl    = ws_ctm    + NROWS;                      // 512 f
    float* ws_tnew   = ws_ftl    + NROWS;                      // 1 f

    cvt_emb_kernel<<<(NROWS * KDIM) / 256, 256, 0, stream>>>(emb, abf);
    target_logit_kernel<<<NROWS, 256, 0, stream>>>(emb, label, kern, ws_target);
    prep_kernel<<<1, NROWS, 0, stream>>>(ws_target, t_in, ws_ctm, ws_ftl, ws_tnew);

    int nstrips = (NCOLS + BN - 1) / BN;   // 1563
    gemm_curricular_kernel<<<nstrips, 512, 0, stream>>>(
        kern, label, abf, ws_ctm, ws_ftl, ws_tnew, out);
}
